// M05SameVQDiffusion_70635032150635
// MI455X (gfx1250) — matrix operand order repaired and checked by reading, not compiled, not run on hardware
//
#include <hip/hip_runtime.h>
#include <hip/hip_bf16.h>

// Problem constants (fixed by the reference).
static constexpr int Bb = 16, Cc = 1024, Tt = 2048, Kk = 4096;
static constexpr int Mrows = Bb * Tt; // 32768 feature vectors

typedef __attribute__((ext_vector_type(16))) __bf16 v16bf;
typedef __attribute__((ext_vector_type(8)))  float  v8f;

__device__ __forceinline__ unsigned short f2bf(float f) {
    unsigned u = __float_as_uint(f);
    u += 0x7FFFu + ((u >> 16) & 1u);   // round-to-nearest-even
    return (unsigned short)(u >> 16);
}

// ---------------------------------------------------------------- kernel 1 --
// centroid fp32 (K,C) -> bf16 copy + csq[k] = ||c_k||^2
__global__ void __launch_bounds__(256)
prep_centroid(const float* __restrict__ cent,
              unsigned short* __restrict__ cbf,
              float* __restrict__ csq) {
    __shared__ float red[256];
    const int k   = blockIdx.x;
    const int tid = threadIdx.x;
    const float* row = cent + (size_t)k * Cc;
    unsigned short* orow = cbf + (size_t)k * Cc;
    float s = 0.f;
    #pragma unroll
    for (int j = 0; j < Cc; j += 256) {
        float v = row[j + tid];
        s += v * v;
        orow[j + tid] = f2bf(v);
    }
    red[tid] = s;
    __syncthreads();
    for (int st = 128; st > 0; st >>= 1) {
        if (tid < st) red[tid] += red[tid + st];
        __syncthreads();
    }
    if (tid == 0) csq[k] = red[0];
}

// ---------------------------------------------------------------- kernel 2 --
// feature (B,C,T) fp32 -> xb (B*T, C) bf16, coalesced via LDS tile transpose
__global__ void __launch_bounds__(256)
transpose_feature(const float* __restrict__ feat,
                  unsigned short* __restrict__ xb) {
    __shared__ float tile[32][33];
    const int t0 = blockIdx.x * 32;
    const int c0 = blockIdx.y * 32;
    const int b  = blockIdx.z;
    const int tx = threadIdx.x, ty = threadIdx.y;
    const float* fb = feat + (size_t)b * Cc * Tt;
    #pragma unroll
    for (int j = 0; j < 4; ++j) {
        int cl = ty + j * 8;
        tile[cl][tx] = fb[(size_t)(c0 + cl) * Tt + t0 + tx];
    }
    __syncthreads();
    #pragma unroll
    for (int j = 0; j < 4; ++j) {
        int tl = ty + j * 8;
        xb[((size_t)b * Tt + t0 + tl) * Cc + c0 + tx] = f2bf(tile[tx][tl]);
    }
}

// ---------------------------------------------------------------- kernel 3 --
// 32 rows per block (two 16-row M-tiles). Each wave owns a stripe of N-tiles
// and computes BOTH M-tiles from one B fragment: halves L2 B-operand traffic
// and issues two independent v_wmma_f32_16x16x32_bf16 per fragment.
// dist = csq[k] - 2 * x.c_k  (||x||^2 constant per row: dropped for argmin).
__global__ void __launch_bounds__(128)
gemm_argmin(const unsigned short* __restrict__ xb,
            const unsigned short* __restrict__ cb,
            const float* __restrict__ csq,
            int* __restrict__ am) {
    __shared__ __align__(16) unsigned short As[32 * Cc]; // 64 KB A slab
    __shared__ float sD[4][32];
    __shared__ int   sI[4][32];

    const int tid  = threadIdx.x;
    const int lane = tid & 31;
    const int wave = tid >> 5;
    const int rowbase = blockIdx.x * 32;

    // Cooperative load of the contiguous 32x1024 bf16 slab (b128 copies).
    {
        const uint4* src = (const uint4*)(xb + (size_t)rowbase * Cc);
        uint4* dst = (uint4*)As;
        for (int i = tid; i < (32 * Cc) / 8; i += 128) dst[i] = src[i];
    }
    __syncthreads();

    const int m    = lane & 15;  // row (A) / column (B,C) within tile
    const int half = lane >> 4;  // K-half selector per ISA 16-bit layouts
    const unsigned short* arow0 = As + m * Cc + half * 8;          // M-tile 0
    const unsigned short* arow1 = As + (16 + m) * Cc + half * 8;   // M-tile 1

    float bd0[8], bd1[8];
    int   bi0[8], bi1[8];
    #pragma unroll
    for (int i = 0; i < 8; ++i) {
        bd0[i] = 3.4e38f; bi0[i] = 0;
        bd1[i] = 3.4e38f; bi1[i] = 0;
    }

    for (int nt = wave; nt < Kk / 16; nt += 4) {
        const int nbase = nt * 16;
        // B column n = centroid row (nbase+m): contiguous bf16 in memory.
        const unsigned short* brow = cb + (size_t)(nbase + m) * Cc + half * 8;
        v8f acc0 = {};
        v8f acc1 = {};
        #pragma unroll 4
        for (int ck = 0; ck < Cc; ck += 32) {
            union { v16bf v; uint4 q[2]; } a0, a1, b;
            // A 16x32 bf16 layout: lanes 0-15 K=[ck..ck+7],[ck+16..ck+23];
            //                      lanes 16-31 K=[ck+8..ck+15],[ck+24..ck+31]
            a0.q[0] = *(const uint4*)(arow0 + ck);
            a0.q[1] = *(const uint4*)(arow0 + ck + 16);
            a1.q[0] = *(const uint4*)(arow1 + ck);
            a1.q[1] = *(const uint4*)(arow1 + ck + 16);
            // B 32x16 bf16 layout: lanes 0-15 K=[ck..ck+7],[ck+16..ck+23]; 16-31 K=[ck+8..ck+15],[ck+24..ck+31]
            b.q[0] = *(const uint4*)(brow + ck);
            b.q[1] = *(const uint4*)(brow + ck + 16);
            __builtin_prefetch(brow + ck + 256, 0, 0);
            acc0 = __builtin_amdgcn_wmma_f32_16x16x32_bf16(
                false, a0.v, false, b.v, (short)0, acc0, false, false);
            acc1 = __builtin_amdgcn_wmma_f32_16x16x32_bf16(
                false, a1.v, false, b.v, (short)0, acc1, false, false);
        }
        const float cs = csq[nbase + m];
        #pragma unroll
        for (int i = 0; i < 8; ++i) {
            float d0 = cs - 2.0f * acc0[i];
            float d1 = cs - 2.0f * acc1[i];
            if (d0 < bd0[i]) { bd0[i] = d0; bi0[i] = nbase + m; } // strict <: first idx wins
            if (d1 < bd1[i]) { bd1[i] = d1; bi1[i] = nbase + m; }
        }
    }

    // Butterfly min-reduce across the 16 lanes of each half-wave
    // (rows i live in lanes 0-15, rows 8+i in lanes 16-31; masks <16 keep halves apart).
    #pragma unroll
    for (int i = 0; i < 8; ++i) {
        float d = bd0[i]; int ix = bi0[i];
        #pragma unroll
        for (int msk = 8; msk >= 1; msk >>= 1) {
            float od = __shfl_xor(d, msk, 32);
            int   oi = __shfl_xor(ix, msk, 32);
            if (od < d || (od == d && oi < ix)) { d = od; ix = oi; }
        }
        bd0[i] = d; bi0[i] = ix;

        d = bd1[i]; ix = bi1[i];
        #pragma unroll
        for (int msk = 8; msk >= 1; msk >>= 1) {
            float od = __shfl_xor(d, msk, 32);
            int   oi = __shfl_xor(ix, msk, 32);
            if (od < d || (od == d && oi < ix)) { d = od; ix = oi; }
        }
        bd1[i] = d; bi1[i] = ix;
    }
    if (lane == 0) {
        #pragma unroll
        for (int i = 0; i < 8; ++i) {
            sD[wave][i]      = bd0[i]; sI[wave][i]      = bi0[i];
            sD[wave][16 + i] = bd1[i]; sI[wave][16 + i] = bi1[i];
        }
    }
    if (lane == 16) {
        #pragma unroll
        for (int i = 0; i < 8; ++i) {
            sD[wave][8 + i]  = bd0[i]; sI[wave][8 + i]  = bi0[i];
            sD[wave][24 + i] = bd1[i]; sI[wave][24 + i] = bi1[i];
        }
    }
    __syncthreads();
    if (tid < 32) {
        float d = sD[0][tid]; int ix = sI[0][tid];
        #pragma unroll
        for (int w = 1; w < 4; ++w) {
            float od = sD[w][tid]; int oi = sI[w][tid];
            if (od < d || (od == d && oi < ix)) { d = od; ix = oi; }
        }
        am[rowbase + tid] = ix;
    }
}

// ---------------------------------------------------------------- kernel 4 --
// out[b,c,t] = feature[b,c,t] - centroid[am[b*T+t]][c]  (fp32, coalesced)
__global__ void __launch_bounds__(256)
residual(const float* __restrict__ feat,
         const float* __restrict__ cent,
         const int* __restrict__ am,
         float* __restrict__ out) {
    const size_t i = (size_t)blockIdx.x * 256 + threadIdx.x;
    const int t = (int)(i & (Tt - 1));
    const int c = (int)((i >> 11) & (Cc - 1));
    const int b = (int)(i >> 21);
    const int idx = am[((size_t)b << 11) + t];
    out[i] = feat[i] - cent[(size_t)idx * Cc + c];
}

// ------------------------------------------------------------------ launch --
extern "C" void kernel_launch(void* const* d_in, const int* in_sizes, int n_in,
                              void* d_out, int out_size, void* d_ws, size_t ws_size,
                              hipStream_t stream) {
    const float* feat = (const float*)d_in[0];
    const float* cent = (const float*)d_in[1];
    float* out = (float*)d_out;

    char* ws = (char*)d_ws;
    size_t off = 0;
    auto carve = [&](size_t bytes) -> void* {
        void* p = ws + off;
        off = (off + bytes + 255) & ~(size_t)255;
        return p;
    };
    unsigned short* cbf = (unsigned short*)carve((size_t)Kk * Cc * 2);     //  8 MB
    unsigned short* xb  = (unsigned short*)carve((size_t)Mrows * Cc * 2);  // 64 MB
    float* csq          = (float*)carve((size_t)Kk * 4);                   // 16 KB
    int*   am           = (int*)carve((size_t)Mrows * 4);                  // 128 KB
    (void)ws_size; (void)in_sizes; (void)n_in; (void)out_size;

    prep_centroid<<<dim3(Kk), dim3(256), 0, stream>>>(cent, cbf, csq);
    transpose_feature<<<dim3(Tt / 32, Cc / 32, Bb), dim3(32, 8), 0, stream>>>(feat, xb);
    gemm_argmin<<<dim3(Mrows / 32), dim3(128), 0, stream>>>(xb, cbf, csq, am);
    const size_t total = (size_t)Bb * Cc * Tt;
    residual<<<dim3((unsigned)(total / 256)), dim3(256), 0, stream>>>(feat, cent, am, out);
}
